// GNN_34729105555468
// MI455X (gfx1250) — compile-verified, matrix operand order checked by reading, not run on hardware
//
#include <hip/hip_runtime.h>
#include <hip/hip_bf16.h>
#include <stdint.h>

// ---------------------------------------------------------------------------
// GNN: 3x sparse mean-aggregate layers + dense 256->10 projection (WMMA).
// Strategy: transpose activations to (feat, batch) so every edge gather /
// dst accumulation is a contiguous, fully-coalesced 1024-float (4KB) row.
// The 82MB transposed layer-0 activation fits in MI455X's 192MB L2, so the
// ~400MB of edge-gather traffic is L2-resident. The SpMM hot loop double-
// buffers rows through LDS with global_load_async_to_lds_b128 (ASYNCcnt).
// Final dense GEMM uses v_wmma_f32_16x16x32_f16 (wave32, 16x16/wave).
// ---------------------------------------------------------------------------

#define BATCH 1024
#define F0 20000
#define N0 5000
#define N1 1000
#define N2 256

typedef __attribute__((ext_vector_type(16))) _Float16 v16h;
typedef __attribute__((ext_vector_type(8)))  float    v8f;

// ---------------- 1) fused transpose + per-feature scale -------------------
// xT[f][b] = data[b][f] * p[f]
__global__ void transpose_scale_kernel(const float* __restrict__ data,
                                       const float* __restrict__ p,
                                       float* __restrict__ xT,
                                       int B, int F) {
  __shared__ float tile[32][33];
  const int f0 = blockIdx.x * 32;
  const int b0 = blockIdx.y * 32;
  for (int i = threadIdx.y; i < 32; i += 8) {
    int b = b0 + i, f = f0 + threadIdx.x;
    tile[i][threadIdx.x] = (b < B && f < F) ? data[(size_t)b * F + f] : 0.0f;
  }
  __syncthreads();
  for (int i = threadIdx.y; i < 32; i += 8) {
    int f = f0 + i, b = b0 + threadIdx.x;
    if (f < F && b < B)
      xT[(size_t)f * B + b] = tile[threadIdx.x][i] * p[f];
  }
}

// ---------------- 2) CSR-by-dst construction -------------------------------
__global__ void hist_kernel(const int* __restrict__ dst, int ne,
                            int* __restrict__ counts) {
  int e = blockIdx.x * 256 + threadIdx.x;
  if (e < ne) atomicAdd(&counts[dst[e]], 1);
}

// single-block exclusive scan (n <= 5000), also primes the scatter cursor
__global__ void exscan_kernel(const int* __restrict__ counts, int n,
                              int* __restrict__ offsets,
                              int* __restrict__ cursor) {
  __shared__ int sm[1024];
  __shared__ int carry;
  if (threadIdx.x == 0) carry = 0;
  __syncthreads();
  for (int base = 0; base < n; base += 1024) {
    int i = base + threadIdx.x;
    int v = (i < n) ? counts[i] : 0;
    sm[threadIdx.x] = v;
    __syncthreads();
    for (int off = 1; off < 1024; off <<= 1) {
      int t = (threadIdx.x >= (unsigned)off) ? sm[threadIdx.x - off] : 0;
      __syncthreads();
      sm[threadIdx.x] += t;
      __syncthreads();
    }
    int excl = sm[threadIdx.x] - v + carry;
    if (i < n) { offsets[i] = excl; cursor[i] = excl; }
    __syncthreads();
    if (threadIdx.x == 0) carry += sm[1023];
    __syncthreads();
  }
  if (threadIdx.x == 0) offsets[n] = carry;
}

__global__ void scatter_kernel(const int* __restrict__ src,
                               const int* __restrict__ dst, int ne,
                               int* __restrict__ cursor,
                               int* __restrict__ sorted) {
  int e = blockIdx.x * 256 + threadIdx.x;
  if (e < ne) {
    int pos = atomicAdd(&cursor[dst[e]], 1);
    sorted[pos] = src[e];
  }
}

// make per-bucket order deterministic (fp-sum order fixed across replays)
__global__ void sort_buckets_kernel(const int* __restrict__ offsets,
                                    int* __restrict__ sorted, int nout) {
  int j = blockIdx.x * 256 + threadIdx.x;
  if (j >= nout) return;
  int s = offsets[j], e = offsets[j + 1];
  for (int i = s + 1; i < e; ++i) {
    int key = sorted[i];
    int k = i - 1;
    while (k >= s && sorted[k] > key) { sorted[k + 1] = sorted[k]; --k; }
    sorted[k + 1] = key;
  }
}

// ---------------- 3) SpMM: per-dst row accumulate, mean, ReLU --------------
// block = 256 threads (8 wave32), one dst row of 1024 floats.
// Async double-buffer: while accumulating row e from LDS, row e+1 streams
// from L2/HBM into the other LDS buffer via global_load_async_to_lds_b128.
// Per-wave ASYNCcnt completes in order, so s_wait_asynccnt 1 guarantees the
// current buffer's 16B chunk landed; __syncthreads covers cross-wave chunks.
template <bool USE_P>
__global__ void spmm_rows_kernel(const float* __restrict__ xT,
                                 const float* __restrict__ p,
                                 const int* __restrict__ offsets,
                                 const int* __restrict__ sorted,
                                 float* __restrict__ yT) {
  __shared__ float buf[2][BATCH];      // 2 x 4KB double buffer
  const int j = blockIdx.x;
  const int t = threadIdx.x;           // 0..255 -> 16B chunk per thread
  const int start = offsets[j];
  const int end   = offsets[j + 1];

  // LDS byte addresses of this thread's 16B chunk in each buffer
  const uint32_t lds0 = (uint32_t)(uintptr_t)(&buf[0][t * 4]);
  const uint32_t lds1 = (uint32_t)(uintptr_t)(&buf[1][t * 4]);

  float4 acc = make_float4(0.f, 0.f, 0.f, 0.f);

  if (start < end) {                   // prime the pipeline with row[start]
    const float* g = xT + (size_t)sorted[start] * BATCH + t * 4;
    asm volatile("global_load_async_to_lds_b128 %0, %1, off"
                 :: "v"(lds0), "v"(g) : "memory");
  }
  for (int e = start; e < end; ++e) {
    const int s = sorted[e];
    if (e + 1 < end) {                 // stream next row into the other buffer
      const float* g = xT + (size_t)sorted[e + 1] * BATCH + t * 4;
      const uint32_t ldsn = ((e + 1 - start) & 1) ? lds1 : lds0;
      asm volatile("global_load_async_to_lds_b128 %0, %1, off"
                   :: "v"(ldsn), "v"(g) : "memory");
      asm volatile("s_wait_asynccnt 0x1" ::: "memory");  // current row done
    } else {
      asm volatile("s_wait_asynccnt 0x0" ::: "memory");
    }
    __syncthreads();                   // all waves' chunks of row e visible

    const float4 v = *(const float4*)(&buf[(e - start) & 1][t * 4]);
    float w = 1.0f;
    if (USE_P) w = p[s];
    acc.x += v.x * w; acc.y += v.y * w;
    acc.z += v.z * w; acc.w += v.w * w;

    __syncthreads();                   // row e consumed before buffer reuse
  }
  const float cnt = (float)(end - start);
  const float inv = 1.0f / fmaxf(cnt, 1.0f);
  float4 o;
  o.x = fmaxf(acc.x * inv, 0.f);
  o.y = fmaxf(acc.y * inv, 0.f);
  o.z = fmaxf(acc.z * inv, 0.f);
  o.w = fmaxf(acc.w * inv, 0.f);
  ((float4*)(yT + (size_t)j * BATCH))[t] = o;
}

// ---------------- 4) final dense GEMM via WMMA -----------------------------
// out(1024,10) = x2(1024,256) @ W(10,256)^T + b; x provided transposed as
// x3T (256 x 1024). One wave per 16(M)x16(N) tile, N padded 10->16.
// Fully unrolled: 8 x v_wmma_f32_16x16x32_f16 with f32 accumulation.
__global__ void final_gemm_wmma_kernel(const float* __restrict__ x3T, // 256x1024
                                       const float* __restrict__ W,   // 10x256
                                       const float* __restrict__ bias,// 10
                                       float* __restrict__ out) {     // 1024x10
  const int lane   = threadIdx.x;      // wave32
  const int m_base = blockIdx.x * 16;  // 64 blocks cover M=1024
  const int m      = lane & 15;
  const int half   = lane >> 4;        // 0: K 0..7,16..23; 1: K 8..15,24..31
  const int khalf  = half * 8;
  const int n      = lane & 15;
  const int nw     = (n < 10) ? n : 9; // clamp to stay in-bounds on W

  v8f c = {};
#pragma unroll
  for (int k0 = 0; k0 < 256; k0 += 32) {
    v16h a, bm;
#pragma unroll
    for (int e = 0; e < 16; ++e) {
      const int pr = e >> 1;
      // 16-bit A 16x32 layout: VGPR r<4 -> K = khalf+2r(+1); r>=4 -> +16
      const int k = k0 + ((pr < 4) ? 0 : 16) + khalf + 2 * (pr & 3) + (e & 1);
      a[e] = (_Float16)x3T[(size_t)k * BATCH + m_base + m];
      float wv = W[nw * 256 + k];
      bm[e] = (n < 10) ? (_Float16)wv : (_Float16)0.0f;
    }
    // 8 args: (neg_a, A, neg_b, B, c_mod, C, reuse_a, reuse_b)
    c = __builtin_amdgcn_wmma_f32_16x16x32_f16(false, a, false, bm,
                                               (short)0, c, false, false);
  }
  if (n < 10) {
    const float bv = bias[n];
#pragma unroll
    for (int r = 0; r < 8; ++r) {
      const int mm = m_base + r + 8 * half;   // C/D: reg r -> M=r (+8 for half)
      out[mm * 10 + n] = c[r] + bv;
    }
  }
}

// ---------------------------------------------------------------------------
static void run_layer(const float* xT, const float* p,
                      const int* src, const int* dst, int ne, int nout,
                      float* yT, int* counts, int* offsets, int* cursor,
                      int* sorted, hipStream_t stream) {
  hipMemsetAsync(counts, 0, (size_t)nout * sizeof(int), stream);
  const int eb = (ne + 255) / 256;
  hist_kernel<<<eb, 256, 0, stream>>>(dst, ne, counts);
  exscan_kernel<<<1, 1024, 0, stream>>>(counts, nout, offsets, cursor);
  scatter_kernel<<<eb, 256, 0, stream>>>(src, dst, ne, cursor, sorted);
  sort_buckets_kernel<<<(nout + 255) / 256, 256, 0, stream>>>(offsets, sorted, nout);
  if (p)
    spmm_rows_kernel<true><<<nout, 256, 0, stream>>>(xT, p, offsets, sorted, yT);
  else
    spmm_rows_kernel<false><<<nout, 256, 0, stream>>>(xT, nullptr, offsets, sorted, yT);
}

extern "C" void kernel_launch(void* const* d_in, const int* in_sizes, int n_in,
                              void* d_out, int out_size, void* d_ws, size_t ws_size,
                              hipStream_t stream) {
  (void)n_in; (void)out_size; (void)ws_size;
  const float* data = (const float*)d_in[0];
  const float* p0   = (const float*)d_in[1];
  const float* p1   = (const float*)d_in[2];
  const float* p2   = (const float*)d_in[3];
  const float* W    = (const float*)d_in[4];
  const float* bias = (const float*)d_in[5];
  const int* src0 = (const int*)d_in[6];
  const int* dst0 = (const int*)d_in[7];
  const int* src1 = (const int*)d_in[8];
  const int* dst1 = (const int*)d_in[9];
  const int* src2 = (const int*)d_in[10];
  const int* dst2 = (const int*)d_in[11];
  const int ne0 = in_sizes[6], ne1 = in_sizes[8], ne2 = in_sizes[10];

  // workspace layout (~108 MB total)
  size_t off = 0;
  char* base = (char*)d_ws;
  auto alloc = [&](size_t bytes) -> void* {
    void* ptr = base + off;
    off += (bytes + 255) & ~(size_t)255;
    return ptr;
  };
  float* xT0  = (float*)alloc((size_t)F0 * BATCH * 4);   // 82 MB (L2-resident)
  float* x1T  = (float*)alloc((size_t)N0 * BATCH * 4);   // 20.5 MB
  float* x2T  = (float*)alloc((size_t)N1 * BATCH * 4);   // 4.1 MB
  float* x3T  = (float*)alloc((size_t)N2 * BATCH * 4);   // 1 MB
  int* counts  = (int*)alloc((size_t)N0 * 4);
  int* offsets = (int*)alloc(((size_t)N0 + 1) * 4);
  int* cursor  = (int*)alloc((size_t)N0 * 4);
  int* sorted  = (int*)alloc((size_t)ne0 * 4);

  // layer 0 input: transpose + fuse p0 scaling (coalesced both directions)
  transpose_scale_kernel<<<dim3((F0 + 31) / 32, (BATCH + 31) / 32),
                           dim3(32, 8), 0, stream>>>(data, p0, xT0, BATCH, F0);

  run_layer(xT0, nullptr, src0, dst0, ne0, N0, x1T,
            counts, offsets, cursor, sorted, stream);      // p0 pre-fused
  run_layer(x1T, p1, src1, dst1, ne1, N1, x2T,
            counts, offsets, cursor, sorted, stream);
  run_layer(x2T, p2, src2, dst2, ne2, N2, x3T,
            counts, offsets, cursor, sorted, stream);

  // dense projection with WMMA (64 tiles of 16x16, one wave32 each)
  final_gemm_wmma_kernel<<<64, 32, 0, stream>>>(x3T, W, bias, (float*)d_out);
}
